// InvertedResidual_pruned_11570641895469
// MI455X (gfx1250) — matrix operand order repaired and checked by reading, not compiled, not run on hardware
//
#include <hip/hip_runtime.h>

typedef _Float16 v16h __attribute__((ext_vector_type(16)));
typedef _Float16 v8h  __attribute__((ext_vector_type(8)));
typedef _Float16 v4h  __attribute__((ext_vector_type(4)));
typedef float    v8f  __attribute__((ext_vector_type(8)));

#define EPSBN 1e-5f
#define CIN   96
#define HH    56
#define WW    56
#define P0C   384
#define P1C   72
#define CHUNK 64      // expanded channels per pass (K granularity 2x32 for WMMA)
#define NCHUNK 6      // 384/64
#define TR    4       // output tile rows
#define TC    16      // output tile cols
#define RPIX  108     // halo region pixels: 6*18
#define RPAD  112     // padded to 7 WMMA M-tiles
#define OPIX  64      // output pixels per tile (4 M-tiles)
#define XSTR  104     // xs row stride in halves: mult of 8 (16B align), bank-spread
#define ESTR  64      // es row stride in halves

// ---- wave32 WMMA fragment loaders (CDNA5 ISA 7.12.2) ----
// A (16x32 f16 MxK): lane l -> row M=mBase+(l&15); halves 0..15 = K g..g+15
// (g = (l>>4)*16). Contiguous in LDS
// when A is stored row-major -> two aligned ds_load_b128 per fragment.
__device__ __forceinline__ v16h loadFragA(const _Float16* s, int mBase,
                                          int stride, int kBase, int lane) {
  const _Float16* row = s + (mBase + (lane & 15)) * stride + kBase + ((lane >> 4) << 4);
  const v8h lo = *(const v8h*)row;          // K offsets +0..7
  const v8h hi = *(const v8h*)(row + 8);    // K offsets +8..15
  return __builtin_shufflevector(lo, hi, 0, 1, 2, 3, 4, 5, 6, 7,
                                 8, 9, 10, 11, 12, 13, 14, 15);
}

// B fragments are staged in register order: frag base + lane*16 halves,
// half h <-> K = (lane>>4)*16 + h, N = lane&15 -> two ds_load_b128.
__device__ __forceinline__ v16h loadFragB(const _Float16* fragBase, int lane) {
  const _Float16* p = fragBase + lane * 16;
  const v8h lo = *(const v8h*)p;
  const v8h hi = *(const v8h*)(p + 8);
  return __builtin_shufflevector(lo, hi, 0, 1, 2, 3, 4, 5, 6, 7,
                                 8, 9, 10, 11, 12, 13, 14, 15);
}

__global__ __launch_bounds__(128, 1) void mbconv_fused(
    const float* __restrict__ x,  const float* __restrict__ wE,
    const float* __restrict__ wD, const float* __restrict__ wP,
    const float* __restrict__ g1, const float* __restrict__ b1,
    const float* __restrict__ m1, const float* __restrict__ v1,
    const float* __restrict__ g2, const float* __restrict__ b2,
    const float* __restrict__ m2, const float* __restrict__ v2,
    const float* __restrict__ g3, const float* __restrict__ b3,
    const float* __restrict__ m3, const float* __restrict__ v3,
    const int* __restrict__ idx,  float* __restrict__ out) {
  // ---- LDS (63,104 B) ----
  __shared__ __align__(16) _Float16 xs[RPAD * XSTR];   // 23296 B  x tile f16 [pix][ic]
  __shared__ __align__(16) _Float16 es[RPAD * ESTR];   // 14336 B  expanded acts; reused as ps[64][64]
  __shared__ __align__(16) _Float16 weF[12 * 512];     // 12288 B  expand B frags [kk*4+nt][lane][16]
  __shared__ __align__(16) _Float16 wpF[10 * 512];     // 10240 B  project B frags [kk*5+nt][lane][16]
  __shared__ __align__(16) _Float16 dwWh[9 * CHUNK];   //  1152 B  dw weights f16 [tap][oc]
  __shared__ __align__(16) _Float16 sc2h[CHUNK], sh2h[CHUNK];  // 256 B folded BN2 (f16)
  __shared__ float sc1[CHUNK], sh1[CHUNK];
  __shared__ float sc3[80], sh3[80];
  __shared__ int   map96[CIN];

  const int tid  = threadIdx.x;
  const int lane = tid & 31;
  const int wave = tid >> 5;               // 4 waves

  const int bid = blockIdx.x;
  const int img = bid / 56;
  const int t   = bid % 56;                // 14 row-tiles x 4 col-tiles
  const int r0  = (t >> 2) * TR;
  const int c0  = (t & 3) * TC;

  const float* xb = x   + (size_t)img * CIN * HH * WW;
  float*       ob = out + (size_t)img * CIN * HH * WW;

  // ---- phase 0: scatter map, BN3 fold, stage x tile as f16 ----
  if (tid < CIN) map96[tid] = -1;
  __syncthreads();
  if (tid < P1C) {
    map96[idx[tid]] = tid;
    const float s = g3[tid] * rsqrtf(v3[tid] + EPSBN);
    sc3[tid] = s;
    sh3[tid] = b3[tid] - m3[tid] * s;
  }
  for (int i = tid; i < RPAD * CIN; i += 128) {
    const int ic = i / RPAD, pix = i % RPAD;
    float v = 0.f;
    if (pix < RPIX) {
      const int gy = r0 + pix / 18 - 1;
      const int gx = c0 + pix % 18 - 1;
      if (gy >= 0 && gy < HH && gx >= 0 && gx < WW)
        v = xb[(ic * HH + gy) * WW + gx];
    }
    xs[pix * XSTR + ic] = (_Float16)v;
  }

  // project accumulators live across chunks (K-chunked accumulation)
  v8f pacc[5];
#pragma unroll
  for (int nt = 0; nt < 5; ++nt) { v8f z = {}; pacc[nt] = z; }

  // ---- chunk loop over expanded channels ----
  for (int cc = 0; cc < NCHUNK; ++cc) {
    const int ch0 = cc * CHUNK;
    __syncthreads();   // previous chunk consumers done (also fences phase-0 stores)

    // expand weights -> B-fragment order. frag f = kk*4 + nt.
    for (int i = tid; i < CIN * CHUNK; i += 128) {
      const int oc = i / CIN, ic = i % CIN;           // ic fastest: coalesced global
      const int kk = ic >> 5, kin = ic & 31;
      const int l = (oc & 15) + (kin & 16);
      weF[((kk << 2) + (oc >> 4)) * 512 + l * 16 + (kin & 15)] =
          (_Float16)wE[(ch0 + oc) * CIN + ic];
    }
    // project weights -> B-fragment order (N padded 72->80). frag f = kk*5 + nt.
    for (int i = tid; i < CHUNK * 80; i += 128) {
      const int n = i / CHUNK, k = i % CHUNK;         // k fastest: coalesced global
      const float v = (n < P1C) ? wP[n * P0C + ch0 + k] : 0.f;
      const int kk = k >> 5, kin = k & 31;
      const int l = (n & 15) + (kin & 16);
      wpF[(kk * 5 + (n >> 4)) * 512 + l * 16 + (kin & 15)] = (_Float16)v;
    }
    // dw weights f16 tap-major so a quad of channels is one ds_load_b64
    for (int i = tid; i < CHUNK * 9; i += 128) {
      const int oc = i % CHUNK, tap = i / CHUNK;
      dwWh[tap * CHUNK + oc] = (_Float16)wD[(ch0 + oc) * 9 + tap];
    }
    if (tid < CHUNK) {
      const int c = ch0 + tid;
      const float s1v = g1[c] * rsqrtf(v1[c] + EPSBN);
      sc1[tid] = s1v; sh1[tid] = b1[c] - m1[c] * s1v;
      const float s2v = g2[c] * rsqrtf(v2[c] + EPSBN);
      sc2h[tid] = (_Float16)s2v;
      sh2h[tid] = (_Float16)(b2[c] - m2[c] * s2v);
    }
    // prefetch next chunk's weights into cache (global_prefetch_b8)
    if (cc + 1 < NCHUNK) {
      const int nch0 = ch0 + CHUNK;
      __builtin_prefetch(wE + (size_t)nch0 * CIN + tid * 48, 0, 1);
      if (tid < P1C) __builtin_prefetch(wP + (size_t)tid * P0C + nch0, 0, 1);
      if (tid < 18) __builtin_prefetch(wD + (size_t)nch0 * 9 + tid * 32, 0, 1);
    }
    __syncthreads();

    // ---- expand GEMM over halo region: M=112 (7 tiles), N=64 (4 tiles), K=96 ----
    for (int mt = wave; mt < 7; mt += 4) {
      const v16h a0 = loadFragA(xs, mt * 16, XSTR, 0,  lane);
      const v16h a1 = loadFragA(xs, mt * 16, XSTR, 32, lane);
      const v16h a2 = loadFragA(xs, mt * 16, XSTR, 64, lane);
#pragma unroll
      for (int nt = 0; nt < 4; ++nt) {
        v8f acc = {};
        acc = __builtin_amdgcn_wmma_f32_16x16x32_f16(
            false, a0, false, loadFragB(weF + (0 + nt) * 512, lane), (short)0, acc, false, false);
        acc = __builtin_amdgcn_wmma_f32_16x16x32_f16(
            false, a1, false, loadFragB(weF + (4 + nt) * 512, lane), (short)0, acc, false, false);
        acc = __builtin_amdgcn_wmma_f32_16x16x32_f16(
            false, a2, false, loadFragB(weF + (8 + nt) * 512, lane), (short)0, acc, false, false);
        const int oc = nt * 16 + (lane & 15);
        const float s = sc1[oc], sh = sh1[oc];
#pragma unroll
        for (int r = 0; r < 8; ++r) {
          const int m = mt * 16 + r + ((lane >> 4) << 3);   // C/D layout row
          const float v = fminf(fmaxf(acc[r] * s + sh, 0.f), 6.f);
          es[m * ESTR + oc] = (_Float16)v;
        }
      }
    }
    __syncthreads();

    // ---- depthwise 3x3 in packed f16 (v_pk_fma_f16): 4 channels x 8 pixels ----
    const int oc0 = (tid & 15) << 2;                 // same quad for all j
    v4h w4[9];
#pragma unroll
    for (int tp = 0; tp < 9; ++tp)
      w4[tp] = *(const v4h*)(dwWh + tp * CHUNK + oc0);     // ds_load_b64
    const v4h s2q = *(const v4h*)(sc2h + oc0);
    const v4h h2q = *(const v4h*)(sh2h + oc0);
    const v4h zero4 = {(_Float16)0, (_Float16)0, (_Float16)0, (_Float16)0};
    const v4h six4  = {(_Float16)6, (_Float16)6, (_Float16)6, (_Float16)6};
    v4h dreg[8];
#pragma unroll
    for (int j = 0; j < 8; ++j) {
      const int opix = (tid >> 4) + j * 8;           // 0..63
      const int ry = opix >> 4, cx = opix & 15;
      v4h a4 = zero4;
#pragma unroll
      for (int dy = 0; dy < 3; ++dy)
#pragma unroll
        for (int dx = 0; dx < 3; ++dx) {
          const int pix = (ry + dy) * 18 + cx + dx;
          const v4h nb = *(const v4h*)(es + pix * ESTR + oc0);   // ds_load_b64
          a4 += nb * w4[dy * 3 + dx];                // 2x v_pk_fma_f16
        }
      v4h r4 = a4 * s2q + h2q;                       // folded BN2
      r4 = __builtin_elementwise_min(__builtin_elementwise_max(r4, zero4), six4);
      dreg[j] = r4;
    }
    __syncthreads();                                  // all halo reads done
#pragma unroll
    for (int j = 0; j < 8; ++j) {                     // ps[opix][oc] reuses es
      const int opix = (tid >> 4) + j * 8;
      *(v4h*)(es + opix * ESTR + oc0) = dreg[j];      // ds_store_b64
    }
    __syncthreads();

    // ---- project partial GEMM: M=64 (1 tile/wave), N=80 (5 tiles), K=64 ----
    const v16h p0 = loadFragA(es, wave * 16, ESTR, 0,  lane);
    const v16h p1 = loadFragA(es, wave * 16, ESTR, 32, lane);
#pragma unroll
    for (int nt = 0; nt < 5; ++nt) {
      pacc[nt] = __builtin_amdgcn_wmma_f32_16x16x32_f16(
          false, p0, false, loadFragB(wpF + (0 + nt) * 512, lane), (short)0, pacc[nt], false, false);
      pacc[nt] = __builtin_amdgcn_wmma_f32_16x16x32_f16(
          false, p1, false, loadFragB(wpF + (5 + nt) * 512, lane), (short)0, pacc[nt], false, false);
    }
  }

  // ---- epilogue ----
  // passthrough: out = x for channels not in idx (disjoint from scatter below)
  for (int i = tid; i < CIN * OPIX; i += 128) {
    const int c = i / OPIX, opix = i % OPIX;
    if (map96[c] >= 0) continue;
    const int gy = r0 + (opix >> 4), gx = c0 + (opix & 15);
    if (gx < WW) {
      const size_t off = (size_t)(c * HH + gy) * WW + gx;
      ob[off] = xb[off];
    }
  }
  // scatter preserved channels: BN3 + residual
#pragma unroll
  for (int nt = 0; nt < 5; ++nt) {
    const int j = nt * 16 + (lane & 15);
    if (j >= P1C) continue;
    const int c = idx[j];
    const float s = sc3[j], sh = sh3[j];
#pragma unroll
    for (int r = 0; r < 8; ++r) {
      const int opix = wave * 16 + r + ((lane >> 4) << 3);
      const int gy = r0 + (opix >> 4), gx = c0 + (opix & 15);
      if (gx < WW) {
        const size_t off = (size_t)(c * HH + gy) * WW + gx;
        ob[off] = xb[off] + pacc[nt][r] * s + sh;
      }
    }
  }
}

extern "C" void kernel_launch(void* const* d_in, const int* in_sizes, int n_in,
                              void* d_out, int out_size, void* d_ws, size_t ws_size,
                              hipStream_t stream) {
  (void)in_sizes; (void)n_in; (void)out_size; (void)d_ws; (void)ws_size;
  const float* x  = (const float*)d_in[0];
  const float* wE = (const float*)d_in[1];
  const float* wD = (const float*)d_in[2];
  const float* wP = (const float*)d_in[3];
  const float* g1 = (const float*)d_in[4];
  const float* b1 = (const float*)d_in[5];
  const float* m1 = (const float*)d_in[6];
  const float* v1 = (const float*)d_in[7];
  const float* g2 = (const float*)d_in[8];
  const float* b2 = (const float*)d_in[9];
  const float* m2 = (const float*)d_in[10];
  const float* v2 = (const float*)d_in[11];
  const float* g3 = (const float*)d_in[12];
  const float* b3 = (const float*)d_in[13];
  const float* m3 = (const float*)d_in[14];
  const float* v3 = (const float*)d_in[15];
  const int*  idx = (const int*)d_in[16];
  float* out = (float*)d_out;

  // 32 images * (14 row-tiles x 4 col-tiles) = 1792 blocks, 128 threads (4 waves)
  mbconv_fused<<<dim3(32 * 56), dim3(128), 0, stream>>>(
      x, wE, wD, wP, g1, b1, m1, v1, g2, b2, m2, v2, g3, b3, m3, v3, idx, out);
}
